// CxNE_OBSOLETE_45732811767855
// MI455X (gfx1250) — matrix operand order repaired and checked by reading, not compiled, not run on hardware
//
#include <hip/hip_runtime.h>
#include <math.h>

typedef __attribute__((ext_vector_type(16))) _Float16 v16h;
typedef __attribute__((ext_vector_type(8)))  float    v8f;

#define GN 50000
#define GE 800000
#define GH 4
#define GC 16
#define GHC 64
#define SLOPE 0.2f
#define BN_EPS 1e-5f
#define MTILES (GN / 16)   // 3125

// ---------------------------------------------------------------------------
// fill
// ---------------------------------------------------------------------------
__global__ void fill_kernel(float* __restrict__ p, float v, int n) {
    int t = blockIdx.x * blockDim.x + threadIdx.x;
    if (t < n) p[t] = v;
}

// ---------------------------------------------------------------------------
// weight prep: f32 [K,Nc] row-major -> f16 [Nc,K] column-major (B-fragment
// becomes two contiguous 16B loads per K-step)
// ---------------------------------------------------------------------------
__global__ void wcast_kernel(const float* __restrict__ W,
                             _Float16* __restrict__ Wt, int K, int Nc) {
    int t = blockIdx.x * blockDim.x + threadIdx.x;
    if (t >= K * Nc) return;
    int k = t / Nc, c = t % Nc;
    Wt[(size_t)c * K + k] = (_Float16)W[t];
}

__device__ __forceinline__ float4 bnrelu4(float4 v, float4 s, float4 h) {
    float4 r;
    r.x = fmaxf(v.x * s.x + h.x, 0.0f);
    r.y = fmaxf(v.y * s.y + h.y, 0.0f);
    r.z = fmaxf(v.z * s.z + h.z, 0.0f);
    r.w = fmaxf(v.w * s.w + h.w, 0.0f);
    return r;
}

// ---------------------------------------------------------------------------
// WMMA GEMM: Out[GN,NC] = preop(A[GN,K]) @ W[K,NC] + bias
//   PRE: a' = max(a*sc[k]+sh[k], 0)  (BN+ReLU fused into operand load)
// One wave per 16x16 tile, 4 waves/block. v_wmma_f32_16x16x32_f16, K-loop
// fully unrolled. A fragment: 4x b128 loads; B fragment: 2x 16B loads.
// ---------------------------------------------------------------------------
template <int K, int NC, bool PRE>
__global__ __launch_bounds__(128)
void gemm_wmma_kernel(const float* __restrict__ A,
                      const _Float16* __restrict__ Wt,
                      const float* __restrict__ bias,
                      const float* __restrict__ scv,
                      const float* __restrict__ shv,
                      float* __restrict__ Out) {
    const int lane = threadIdx.x & 31;
    const int tile = blockIdx.x * 4 + (threadIdx.x >> 5);
    if (tile >= MTILES) return;                 // wave-uniform exit (EXEC ok)
    const int half = lane >> 4;
    const int l15  = lane & 15;
    const int r0   = tile * 16;
    const int col  = blockIdx.y * 16 + l15;

    // lane<16 holds K-offsets {0..7,16..23}; lane>=16 holds {8..15,24..31}
    const float*    __restrict__ Arow = A + (size_t)(r0 + l15) * K + 8 * half;
    const _Float16* __restrict__ Bcol = Wt + (size_t)col * K + 8 * half;

    v8f acc = {};
#pragma unroll
    for (int kk = 0; kk < K; kk += 32) {
        float4 x0 = *(const float4*)(Arow + kk);
        float4 x1 = *(const float4*)(Arow + kk + 4);
        float4 y0 = *(const float4*)(Arow + kk + 16);
        float4 y1 = *(const float4*)(Arow + kk + 20);
        if (PRE) {
            const float* scp = scv + kk + 8 * half;
            const float* shp = shv + kk + 8 * half;
            x0 = bnrelu4(x0, *(const float4*)(scp),      *(const float4*)(shp));
            x1 = bnrelu4(x1, *(const float4*)(scp + 4),  *(const float4*)(shp + 4));
            y0 = bnrelu4(y0, *(const float4*)(scp + 16), *(const float4*)(shp + 16));
            y1 = bnrelu4(y1, *(const float4*)(scp + 20), *(const float4*)(shp + 20));
        }
        v16h a;
        a[0]  = (_Float16)x0.x; a[1]  = (_Float16)x0.y;
        a[2]  = (_Float16)x0.z; a[3]  = (_Float16)x0.w;
        a[4]  = (_Float16)x1.x; a[5]  = (_Float16)x1.y;
        a[6]  = (_Float16)x1.z; a[7]  = (_Float16)x1.w;
        a[8]  = (_Float16)y0.x; a[9]  = (_Float16)y0.y;
        a[10] = (_Float16)y0.z; a[11] = (_Float16)y0.w;
        a[12] = (_Float16)y1.x; a[13] = (_Float16)y1.y;
        a[14] = (_Float16)y1.z; a[15] = (_Float16)y1.w;

        v16h b;                                 // same K order as A: {0..7,16..23}+8*half
#pragma unroll
        for (int s = 0; s < 8; s++) { b[s] = Bcol[kk + s]; b[8 + s] = Bcol[kk + 16 + s]; }

        acc = __builtin_amdgcn_wmma_f32_16x16x32_f16(
            false, a, false, b, (short)0, acc, false, false);
    }

    const float bv = bias[col];
#pragma unroll
    for (int v = 0; v < 8; v++) {
        int orow = r0 + v + 8 * half;           // D layout: VGPR v -> rows v/v+8
        Out[(size_t)orow * NC + col] = acc[v] + bv;
    }
}

// ---------------------------------------------------------------------------
// Column stats -> fused BN scale/shift: sc = g*rsqrt(var+eps), sh = bt - m*sc
// ---------------------------------------------------------------------------
__global__ void colstats_kernel(const float* __restrict__ X, int M, int C,
                                const float* __restrict__ g,
                                const float* __restrict__ bt,
                                float* __restrict__ sc, float* __restrict__ sh) {
    __shared__ float s1[256], s2[256];
    int c = blockIdx.x;
    float sum = 0.0f, sq = 0.0f;
    for (int r = threadIdx.x; r < M; r += blockDim.x) {
        float v = X[(size_t)r * C + c];
        sum += v; sq += v * v;
    }
    s1[threadIdx.x] = sum; s2[threadIdx.x] = sq;
    __syncthreads();
    for (int o = 128; o > 0; o >>= 1) {
        if ((int)threadIdx.x < o) {
            s1[threadIdx.x] += s1[threadIdx.x + o];
            s2[threadIdx.x] += s2[threadIdx.x + o];
        }
        __syncthreads();
    }
    if (threadIdx.x == 0) {
        float mean = s1[0] / (float)M;
        float var  = s2[0] / (float)M - mean * mean;   // biased, like torch BN
        float scale = g[c] * rsqrtf(var + BN_EPS);
        sc[c] = scale;
        sh[c] = bt[c] - mean * scale;
    }
}

// ---------------------------------------------------------------------------
// Self-loop attr: per-dst mean of edge_weight
// ---------------------------------------------------------------------------
__global__ void loopattr_edge_kernel(const int* __restrict__ ei,
                                     const float* __restrict__ ew,
                                     float* __restrict__ lasum,
                                     float* __restrict__ cnt) {
    int e = blockIdx.x * blockDim.x + threadIdx.x;
    if (e >= GE) return;
    int d = ei[GE + e];
    atomicAdd(&lasum[d], ew[e]);
    atomicAdd(&cnt[d], 1.0f);
}

__global__ void loopattr_final_kernel(float* __restrict__ la,
                                      const float* __restrict__ cnt) {
    int n = blockIdx.x * blockDim.x + threadIdx.x;
    if (n >= GN) return;
    float c = cnt[n];
    la[n] = la[n] / (c < 1.0f ? 1.0f : c);
}

// ---------------------------------------------------------------------------
// ordered-int float atomic max (m initialized to -3e38)
// ---------------------------------------------------------------------------
__device__ __forceinline__ void atomicMaxF(float* addr, float val) {
    if (val >= 0.0f)
        atomicMax((int*)addr, __float_as_int(val));
    else
        atomicMin((unsigned int*)addr, (unsigned int)__float_as_int(val));
}

// ---------------------------------------------------------------------------
// Edge pass A: logits[e,h] = sum_c att[h,c]*leakyrelu(xl[src]+xr[dst]+ea*we),
// atomicMax into m[dst,h]. One thread per (edge, head); self-loops appended.
// ---------------------------------------------------------------------------
__global__ void edge_logits_kernel(const int* __restrict__ ei,
                                   const float* __restrict__ ew,
                                   const float* __restrict__ la,
                                   const float* __restrict__ xl,
                                   const float* __restrict__ xr,
                                   const float* __restrict__ we,
                                   const float* __restrict__ att,
                                   float* __restrict__ logits,
                                   float* __restrict__ mmax) {
    int t = blockIdx.x * blockDim.x + threadIdx.x;
    const int total = (GE + GN) * GH;
    if (t >= total) return;
    int e = t >> 2, hd = t & 3;
    int s, d; float ea;
    if (e < GE) { s = ei[e]; d = ei[GE + e]; ea = ew[e]; }
    else        { s = e - GE; d = s;          ea = la[s]; }

    const float4* __restrict__ xls = (const float4*)(xl + (size_t)s * GHC + hd * GC);
    const float4* __restrict__ xrd = (const float4*)(xr + (size_t)d * GHC + hd * GC);
    const float4* __restrict__ wee = (const float4*)(we + hd * GC);
    const float4* __restrict__ at4 = (const float4*)(att + hd * GC);

    float acc = 0.0f;
#pragma unroll
    for (int q = 0; q < 4; q++) {
        float4 a = xls[q], b = xrd[q], w = wee[q], av = at4[q];
        float v;
        v = a.x + b.x + ea * w.x; v = v > 0.0f ? v : v * SLOPE; acc += av.x * v;
        v = a.y + b.y + ea * w.y; v = v > 0.0f ? v : v * SLOPE; acc += av.y * v;
        v = a.z + b.z + ea * w.z; v = v > 0.0f ? v : v * SLOPE; acc += av.z * v;
        v = a.w + b.w + ea * w.w; v = v > 0.0f ? v : v * SLOPE; acc += av.w * v;
    }
    logits[t] = acc;
    atomicMaxF(&mmax[d * GH + hd], acc);
}

// ---------------------------------------------------------------------------
// Edge pass B: ex = exp(logit - m[dst,h]); s[dst,h]+=ex; acc[dst,:]+=ex*xl[src]
// (normalization by s deferred to node finalize)
// ---------------------------------------------------------------------------
__global__ void edge_accum_kernel(const int* __restrict__ ei,
                                  const float* __restrict__ logits,
                                  const float* __restrict__ mmax,
                                  const float* __restrict__ xl,
                                  float* __restrict__ ssum,
                                  float* __restrict__ acc) {
    int t = blockIdx.x * blockDim.x + threadIdx.x;
    const int total = (GE + GN) * GH;
    if (t >= total) return;
    int e = t >> 2, hd = t & 3;
    int s, d;
    if (e < GE) { s = ei[e]; d = ei[GE + e]; }
    else        { s = e - GE; d = s; }

    float ex = __expf(logits[t] - mmax[d * GH + hd]);
    atomicAdd(&ssum[d * GH + hd], ex);

    const float4* __restrict__ xls = (const float4*)(xl + (size_t)s * GHC + hd * GC);
    float* __restrict__ ap = acc + (size_t)d * GHC + hd * GC;
#pragma unroll
    for (int q = 0; q < 4; q++) {
        float4 a = xls[q];
        atomicAdd(&ap[4 * q + 0], ex * a.x);
        atomicAdd(&ap[4 * q + 1], ex * a.y);
        atomicAdd(&ap[4 * q + 2], ex * a.z);
        atomicAdd(&ap[4 * q + 3], ex * a.w);
    }
}

// ---------------------------------------------------------------------------
// Node finalize (in place): h[n,hc] = acc[n,hc]/s[n,h] + bias[hc]
// ---------------------------------------------------------------------------
__global__ void node_final_kernel(float* __restrict__ acc,
                                  const float* __restrict__ ssum,
                                  const float* __restrict__ bias) {
    int t = blockIdx.x * blockDim.x + threadIdx.x;
    if (t >= GN * GHC) return;
    int n = t >> 6, hc = t & 63, hd = hc >> 4;
    acc[t] = acc[t] / ssum[n * GH + hd] + bias[hc];
}

// ---------------------------------------------------------------------------
// launch helpers
// ---------------------------------------------------------------------------
static void launch_gemm(const float* A, const float* W, _Float16* wt,
                        const float* bias, const float* sc, const float* sh,
                        int preop, float* Out, int K, int Nc, hipStream_t st) {
    int n = K * Nc;
    wcast_kernel<<<(n + 255) / 256, 256, 0, st>>>(W, wt, K, Nc);
    dim3 grid((MTILES + 3) / 4, Nc / 16);
#define GEMM_CASE(KK, NN)                                                      \
    if (K == KK && Nc == NN) {                                                 \
        if (preop)                                                             \
            gemm_wmma_kernel<KK, NN, true><<<grid, 128, 0, st>>>(              \
                A, wt, bias, sc, sh, Out);                                     \
        else                                                                   \
            gemm_wmma_kernel<KK, NN, false><<<grid, 128, 0, st>>>(             \
                A, wt, bias, sc, sh, Out);                                     \
        return;                                                                \
    }
    GEMM_CASE(256, 128)
    GEMM_CASE(128, 64)
    GEMM_CASE(64, 64)
    GEMM_CASE(64, 32)
#undef GEMM_CASE
}

extern "C" void kernel_launch(void* const* d_in, const int* in_sizes, int n_in,
                              void* d_out, int out_size, void* d_ws, size_t ws_size,
                              hipStream_t stream) {
    const float* x       = (const float*)d_in[0];
    const int*   ei      = (const int*)  d_in[1];
    const float* ew      = (const float*)d_in[2];
    const float* enc_w0  = (const float*)d_in[3];
    const float* enc_b0  = (const float*)d_in[4];
    const float* enc_g0  = (const float*)d_in[5];
    const float* enc_bt0 = (const float*)d_in[6];
    const float* enc_w1  = (const float*)d_in[7];
    const float* enc_b1  = (const float*)d_in[8];
    const float* g_wl[2] = {(const float*)d_in[9],  (const float*)d_in[16]};
    const float* g_bl[2] = {(const float*)d_in[10], (const float*)d_in[17]};
    const float* g_wr[2] = {(const float*)d_in[11], (const float*)d_in[18]};
    const float* g_br[2] = {(const float*)d_in[12], (const float*)d_in[19]};
    const float* g_we[2] = {(const float*)d_in[13], (const float*)d_in[20]};
    const float* g_at[2] = {(const float*)d_in[14], (const float*)d_in[21]};
    const float* g_b [2] = {(const float*)d_in[15], (const float*)d_in[22]};
    const float* bn_g    = (const float*)d_in[23];
    const float* bn_b    = (const float*)d_in[24];
    const float* dec_w0  = (const float*)d_in[25];
    const float* dec_b0  = (const float*)d_in[26];
    const float* dec_g0  = (const float*)d_in[27];
    const float* dec_bt0 = (const float*)d_in[28];
    const float* dec_w1  = (const float*)d_in[29];
    const float* dec_b1  = (const float*)d_in[30];
    float* out = (float*)d_out;

    // workspace arena (floats)
    float* ws   = (float*)d_ws;
    float* pre0 = ws;                 // 50000*128 (later reused as decoder pre)
    float* h0   = ws + 6400000;       // 50000*64
    float* acc  = ws + 9600000;       // 50000*64 (GAT accum -> h1 -> h2, in place)
    float* xl   = ws + 12800000;      // 50000*64
    float* xr   = ws + 16000000;      // 50000*64
    float* lg   = ws + 19200000;      // (E+N)*4
    float* mm   = ws + 22600000;      // N*4
    float* ssum = ws + 22800000;      // N*4
    float* cnt  = ws + 23000000;      // N
    float* la   = ws + 23050000;      // N
    float* sc   = ws + 23100000;      // 128
    float* sh   = ws + 23100128;      // 128
    _Float16* wt = (_Float16*)(ws + 23100256);  // 32768 halfs (f16 W^T scratch)

    const int EN4   = (GE + GN) * GH;
    const int gEN4  = (EN4 + 255) / 256;
    const int gE    = (GE + 255) / 256;
    const int gN    = (GN + 255) / 256;
    const int gNH   = (GN * GH + 255) / 256;
    const int gNHC  = (GN * GHC + 255) / 256;

    // ---------------- encoder ----------------
    launch_gemm(x, enc_w0, wt, enc_b0, nullptr, nullptr, 0, pre0, 256, 128, stream);
    colstats_kernel<<<128, 256, 0, stream>>>(pre0, GN, 128, enc_g0, enc_bt0, sc, sh);
    launch_gemm(pre0, enc_w1, wt, enc_b1, sc, sh, 1, h0, 128, 64, stream);

    // ---------------- self-loop mean edge attr (shared by both GAT layers) --
    fill_kernel<<<gN, 256, 0, stream>>>(la, 0.0f, GN);
    fill_kernel<<<gN, 256, 0, stream>>>(cnt, 0.0f, GN);
    loopattr_edge_kernel<<<gE, 256, 0, stream>>>(ei, ew, la, cnt);
    loopattr_final_kernel<<<gN, 256, 0, stream>>>(la, cnt);

    // ---------------- GAT layers ----------------
    const float* hin = h0;
    for (int L = 0; L < 2; L++) {
        int preop = (L == 1);   // layer 1 consumes relu(bn(h1)) via fused pre-op
        launch_gemm(hin, g_wl[L], wt, g_bl[L], sc, sh, preop, xl, 64, 64, stream);
        launch_gemm(hin, g_wr[L], wt, g_br[L], sc, sh, preop, xr, 64, 64, stream);

        fill_kernel<<<gNH,  256, 0, stream>>>(mm, -3.0e38f, GN * GH);
        fill_kernel<<<gNH,  256, 0, stream>>>(ssum, 0.0f,   GN * GH);
        fill_kernel<<<gNHC, 256, 0, stream>>>(acc,  0.0f,   GN * GHC);

        edge_logits_kernel<<<gEN4, 256, 0, stream>>>(ei, ew, la, xl, xr,
                                                     g_we[L], g_at[L], lg, mm);
        edge_accum_kernel<<<gEN4, 256, 0, stream>>>(ei, lg, mm, xl, ssum, acc);
        node_final_kernel<<<gNHC, 256, 0, stream>>>(acc, ssum, g_b[L]);

        if (L == 0) {
            // shared BN stats for GAT1 input (fused into its GEMM pre-op)
            colstats_kernel<<<64, 256, 0, stream>>>(acc, GN, 64, bn_g, bn_b, sc, sh);
            hin = acc;   // h1 lives in acc; fully consumed by the two GEMMs
                         // above before acc is re-zeroed
        }
    }

    // ---------------- decoder ----------------
    launch_gemm(acc, dec_w0, wt, dec_b0, nullptr, nullptr, 0, pre0, 64, 64, stream);
    colstats_kernel<<<64, 256, 0, stream>>>(pre0, GN, 64, dec_g0, dec_bt0, sc, sh);
    launch_gemm(pre0, dec_w1, wt, dec_b1, sc, sh, 1, out, 64, 32, stream);
}
